// convlstm_68779606278797
// MI455X (gfx1250) — matrix operand order repaired and checked by reading, not compiled, not run on hardware
//
#include <hip/hip_runtime.h>

// ---------------------------------------------------------------------------
// Types for CDNA5 WMMA (wave32): bf16 A/B fragments (16 x bf16), f32 C/D (8 x f32)
// ---------------------------------------------------------------------------
typedef __attribute__((ext_vector_type(16))) __bf16         bf16x16;
typedef __attribute__((ext_vector_type(16))) unsigned short u16x16;
typedef __attribute__((ext_vector_type(8)))  unsigned short u16x8;
typedef __attribute__((ext_vector_type(8)))  float          f32x8;

// Problem constants
#define TT   16
#define BB   8
#define CIN0 16
#define HIDC 32
#define HH   64
#define WW   64
#define HW   (HH * WW)          // 4096
#define KPAD0 448               // 48*9 = 432 -> pad to 448
#define KORG0 432
#define KPAD1 576               // 64*9 = 576 (already mult of 32)
#define KORG1 576

__device__ __forceinline__ float sigm_f(float x) {
    return 1.0f / (1.0f + __expf(-x));
}

__device__ __forceinline__ float tanh_f(float x) {
#if __has_builtin(__builtin_amdgcn_tanhf)
    return __builtin_amdgcn_tanhf(x);      // gfx1250 TRANS op
#else
    return tanhf(x);
#endif
}

__device__ __forceinline__ bf16x16 combine16(u16x8 lo, u16x8 hi) {
    u16x16 r;
#pragma unroll
    for (int i = 0; i < 8; ++i) { r[i] = lo[i]; r[i + 8] = hi[i]; }
    return __builtin_bit_cast(bf16x16, r);
}

// ---------------------------------------------------------------------------
// Weight prep: fp32 [128][Korig] -> bf16 [128][Kpad] (zero padded K tail).
// Row-major [N][K]: per-lane B-fragment gather is a contiguous K run per column.
// ---------------------------------------------------------------------------
__global__ void __launch_bounds__(256)
convlstm_prep_weights(const float* __restrict__ W0, const float* __restrict__ W1,
                      unsigned short* __restrict__ wbf0, unsigned short* __restrict__ wbf1) {
    int idx = blockIdx.x * blockDim.x + threadIdx.x;
    const int N0 = 128 * KPAD0;
    const int N1 = 128 * KPAD1;
    if (idx < N0) {
        int n = idx / KPAD0, k = idx - n * KPAD0;
        float v = (k < KORG0) ? W0[n * KORG0 + k] : 0.0f;
        wbf0[idx] = __builtin_bit_cast(unsigned short, (__bf16)v);
    } else if (idx < N0 + N1) {
        int j = idx - N0;
        int n = j / KPAD1, k = j - n * KPAD1;
        wbf1[j] = __builtin_bit_cast(unsigned short, (__bf16)W1[n * KORG1 + k]);
    }
}

// ---------------------------------------------------------------------------
// Fused ConvLSTM step: implicit-GEMM conv (bf16 WMMA, f32 acc) + gate math.
// Block = 128 threads (4 waves). Block -> (b, y); M-tile = 64 x-positions,
// N = 128 gate channels. Wave w owns rows [16w, 16w+16).
//
// Phase 1: stage whole im2col A tile [64][KPAD] bf16 in LDS (one barrier).
//          Loop over (c, ky): one coalesced row read feeds 3 kx scatter stores.
// Phase 2: barrier-free K loop: ds_load_b128 A-fragments + global b128
//          B-fragments -> 8 WMMAs per 32-wide K chunk per wave.
// Phase 3: LSTM pointwise in registers; h/c routed through padded LDS planes
//          so all global h/c traffic is coalesced float4.
// ---------------------------------------------------------------------------
template <int CIN, int KPAD>
__global__ void __launch_bounds__(128)
convlstm_step(const float* __restrict__ xt,
              const float* __restrict__ hprev, const float* __restrict__ cprev,
              const unsigned short* __restrict__ wbf,
              const float* __restrict__ bias,
              float* __restrict__ hout, float* __restrict__ cout,
              float* __restrict__ hdup, int first) {
    constexpr int CTOT  = CIN + HIDC;
    constexpr int KORIG = CTOT * 9;
    constexpr int PADK  = KPAD - KORIG;

    __shared__ __align__(16) unsigned short ash[64 * KPAD];   // A tile, bf16

    const int tid  = threadIdx.x;
    const int lane = tid & 31;
    const int wave = tid >> 5;
    const int bid  = blockIdx.x;              // [0, 512)
    const int b    = bid >> 6;
    const int y    = bid & 63;

    // ---------------- Phase 1: stage im2col A tile ----------------
    {
        // K-padding tail -> 0
        if constexpr (PADK > 0) {
            for (int e = tid; e < 64 * PADK; e += 128) {
                int row = e / PADK;
                int kk  = e - row * PADK;
                ash[row * KPAD + KORIG + kk] = 0;
            }
        }
        // SAME-pad corner cells never touched by the kx-scatter (disjoint, no race)
        for (int j = tid; j < CTOT * 3; j += 128) {   // j = c*3 + ky
            ash[0 * KPAD + j * 3 + 0]  = 0;           // x=0,  kx=0 -> in[yy][-1]
            ash[63 * KPAD + j * 3 + 2] = 0;           // x=63, kx=2 -> in[yy][64]
        }
        const int p  = tid >> 6;                      // half-block 0/1
        const int xx = tid & 63;                      // source column
        for (int c = p; c < CTOT; c += 2) {
            const float* plane;
            if (c < CIN)      plane = xt + (size_t)(b * CIN + c) * HW;
            else if (!first)  plane = hprev + (size_t)(b * HIDC + (c - CIN)) * HW;
            else              plane = nullptr;        // h0 = 0
            const int kb = c * 9;
#pragma unroll
            for (int ky = 0; ky < 3; ++ky) {
                int yy = y + ky - 1;
                float v = 0.0f;
                if (plane && (unsigned)yy < (unsigned)HH) v = plane[yy * WW + xx];
                unsigned short bv = __builtin_bit_cast(unsigned short, (__bf16)v);
#pragma unroll
                for (int kx = 0; kx < 3; ++kx) {
                    int xr = xx - kx + 1;             // dest row (x position)
                    if ((unsigned)xr < (unsigned)WW)
                        ash[xr * KPAD + kb + ky * 3 + kx] = bv;
                }
            }
        }
    }
    __syncthreads();

    // ---------------- Phase 2: barrier-free WMMA K loop ----------------
    // A (16x32, 16-bit) lane layout: lanes 0-15: K 0-7 & 16-23 ; lanes 16-31: K 8-15 & 24-31.
    // B (32x16, 16-bit) lane layout (matches A): lanes 0-15: K 0-7 & 16-23 ;
    //                                            lanes 16-31: K 8-15 & 24-31.
    const int k0a  = (lane < 16) ? 0 : 8;
    const int k0b  = (lane < 16) ? 0 : 8;
    const int mrow = (wave << 4) + (lane & 15);
    const int ncol = lane & 15;

    f32x8 acc[8];
#pragma unroll
    for (int i = 0; i < 8; ++i) acc[i] = (f32x8)0.0f;

#pragma unroll 2
    for (int kc = 0; kc < KPAD; kc += 32) {
        u16x8 alo = *(const u16x8*)&ash[mrow * KPAD + kc + k0a];
        u16x8 ahi = *(const u16x8*)&ash[mrow * KPAD + kc + k0a + 16];
        bf16x16 afrag = combine16(alo, ahi);
#pragma unroll
        for (int nt = 0; nt < 8; ++nt) {
            int n = (nt << 4) + ncol;
            const unsigned short* wp = wbf + (size_t)n * KPAD + kc + k0b;
            u16x8 blo = *(const u16x8*)wp;        // K k0b .. k0b+7
            u16x8 bhi = *(const u16x8*)(wp + 16); // K k0b+16 .. k0b+23
            bf16x16 bfrag = combine16(blo, bhi);
            acc[nt] = __builtin_amdgcn_wmma_f32_16x16x32_bf16(
                false, afrag, false, bfrag, (short)0, acc[nt], false, false);
        }
    }
    __syncthreads();    // A tile dead; LDS reused for h/c planes

    // ---------------- Phase 3: LSTM pointwise + coalesced I/O ----------------
    constexpr int PST = 68;                           // plane row stride (pad: no bank conflict)
    float* hpl = (float*)ash;                         // [32][68]
    float* cpl = hpl + HIDC * PST;                    // [32][68]

    const int chv = tid >> 2;                         // 0..31 (cooperative rows)
    const int x0  = (tid & 3) << 4;                   // 0/16/32/48
    const size_t gbase = ((size_t)(b * HIDC + chv) * HH + y) * WW + x0;

    if (!first) {                                     // stage c_prev coalesced
        const float4* src = (const float4*)(cprev + gbase);
        float4* dst = (float4*)(cpl + chv * PST + x0);
#pragma unroll
        for (int i = 0; i < 4; ++i) dst[i] = src[i];
    }
    __syncthreads();

    // C/D layout: VGPR r -> row r (lanes 0-15) / r+8 (lanes 16-31); col = lane%16.
    // Gate channels: i -> acc 0,1 ; f -> 2,3 ; o -> 4,5 ; g -> 6,7.
    const int rofs = (lane & 16) ? 8 : 0;
#pragma unroll
    for (int nt = 0; nt < 2; ++nt) {
        int ch = (nt << 4) + ncol;                    // hidden channel [0,32)
        float bi = bias[ch];
        float bf = bias[32 + ch];
        float bo = bias[64 + ch];
        float bg = bias[96 + ch];
#pragma unroll
        for (int r = 0; r < 8; ++r) {
            int xpos = (wave << 4) + r + rofs;
            float gi = acc[nt    ][r] + bi;
            float gf = acc[nt + 2][r] + bf;
            float go = acc[nt + 4][r] + bo;
            float gg = acc[nt + 6][r] + bg;
            int poff = ch * PST + xpos;
            float cold = first ? 0.0f : cpl[poff];
            float cn = sigm_f(gf) * cold + sigm_f(gi) * tanh_f(gg);
            float hn = sigm_f(go) * tanh_f(cn);
            cpl[poff] = cn;                           // each (ch,x) owned by one lane
            hpl[poff] = hn;
        }
    }
    __syncthreads();

    {   // coalesced float4 stores
        const float4* hs = (const float4*)(hpl + chv * PST + x0);
        const float4* cs = (const float4*)(cpl + chv * PST + x0);
        float4* ho = (float4*)(hout + gbase);
        float4* co = (float4*)(cout + gbase);
#pragma unroll
        for (int i = 0; i < 4; ++i) { ho[i] = hs[i]; co[i] = cs[i]; }
        if (hdup) {
            float4* hd = (float4*)(hdup + gbase);
#pragma unroll
            for (int i = 0; i < 4; ++i) hd[i] = hs[i];
        }
    }
}

// ---------------------------------------------------------------------------
// Host launcher
// ---------------------------------------------------------------------------
extern "C" void kernel_launch(void* const* d_in, const int* in_sizes, int n_in,
                              void* d_out, int out_size, void* d_ws, size_t ws_size,
                              hipStream_t stream) {
    const float* x  = (const float*)d_in[0];   // [16,8,16,64,64]
    const float* W0 = (const float*)d_in[1];   // [128,48,3,3]
    const float* b0 = (const float*)d_in[2];   // [128]
    const float* W1 = (const float*)d_in[3];   // [128,64,3,3]
    const float* b1 = (const float*)d_in[4];   // [128]

    float* out = (float*)d_out;
    const size_t S1 = (size_t)TT * BB * HIDC * HW;   // 16,777,216
    const size_t F  = (size_t)BB * HIDC * HW;        //  1,048,576
    float* hseq0 = out;
    float* hseq1 = out + S1;
    float* h0T   = out + 2 * S1;
    float* c0T   = h0T + F;
    float* h1T   = c0T + F;
    float* c1T   = h1T + F;

    // Workspace layout
    unsigned short* wbf0 = (unsigned short*)d_ws;                     // 128*448*2 B
    unsigned short* wbf1 = wbf0 + 128 * KPAD0;                        // 128*576*2 B
    char* wsf = (char*)(wbf1 + 128 * KPAD1);                          // ends at 262144 B
    float* c0buf = (float*)wsf;                                       // 4 MB
    float* c1buf = c0buf + F;                                         // 4 MB

    // bf16 weight prep (deterministic, every call)
    {
        const int total = 128 * KPAD0 + 128 * KPAD1;
        convlstm_prep_weights<<<(total + 255) / 256, 256, 0, stream>>>(W0, W1, wbf0, wbf1);
    }

    const size_t xstep0 = (size_t)BB * CIN0 * HW;   // per-t stride of x
    const size_t hstep  = (size_t)BB * HIDC * HW;   // per-t stride of h_seq

    // Layer 0
    for (int t = 0; t < TT; ++t) {
        const float* xt = x + (size_t)t * xstep0;
        const float* hp = (t == 0) ? nullptr : hseq0 + (size_t)(t - 1) * hstep;
        const float* cp = (t == 0) ? nullptr : c0buf;
        float* ho = hseq0 + (size_t)t * hstep;
        float* co = (t == TT - 1) ? c0T : c0buf;
        float* hd = (t == TT - 1) ? h0T : nullptr;
        convlstm_step<CIN0, KPAD0><<<BB * HH, 128, 0, stream>>>(
            xt, hp, cp, wbf0, b0, ho, co, hd, (t == 0) ? 1 : 0);
    }

    // Layer 1 (input = h_seq0)
    for (int t = 0; t < TT; ++t) {
        const float* xt = hseq0 + (size_t)t * hstep;
        const float* hp = (t == 0) ? nullptr : hseq1 + (size_t)(t - 1) * hstep;
        const float* cp = (t == 0) ? nullptr : c1buf;
        float* ho = hseq1 + (size_t)t * hstep;
        float* co = (t == TT - 1) ? c1T : c1buf;
        float* hd = (t == TT - 1) ? h1T : nullptr;
        convlstm_step<HIDC, KPAD1><<<BB * HH, 128, 0, stream>>>(
            xt, hp, cp, wbf1, b1, ho, co, hd, (t == 0) ? 1 : 0);
    }
}
